// TimeBlock_69200513073657
// MI455X (gfx1250) — matrix operand order repaired and checked by reading, not compiled, not run on hardware
//
#include <hip/hip_runtime.h>
#include <hip/hip_bf16.h>

// ---------------------------------------------------------------------------
// Problem constants (from reference)
// ---------------------------------------------------------------------------
constexpr int B_      = 32;
constexpr int L_      = 512;
constexpr int N_      = 2048;
constexpr int HID_    = 128;
constexpr int CHUNK_  = 64;
constexpr int NCHNK_  = L_ / CHUNK_;          // 8
constexpr int NSEL_   = 153;                  // floor(0.3*512)
constexpr int G_      = NSEL_ + CHUNK_ + L_;  // 729
constexpr int GP_     = 736;                  // 23*32, K padded for WMMA
constexpr int GSTRIDE = 744;                  // f16 elems; 372 dwords -> conflict-free b128
constexpr int HSTRIDE = 136;                  // f16 elems; 68 dwords  -> conflict-free b128
constexpr int ROWS_   = 64;                   // n-rows per block (4 WMMA M-tiles)
constexpr int MT_     = ROWS_ / 16;           // 4 M-tiles per wave
constexpr int KT1_    = GP_ / 32;             // 23 k-steps in GEMM1

typedef __attribute__((ext_vector_type(16))) _Float16 v16h;
typedef __attribute__((ext_vector_type(8)))  _Float16 v8h;
typedef __attribute__((ext_vector_type(8)))  float    v8f;

// ---------------------------------------------------------------------------
// Prep: w1 (729x128 f32) -> w1t f16 [128][736] (transposed, K zero-padded)
//       w2 (128x512 f32) -> w2t f16 [512][128] (transposed)
// Weights then live in L2 (192 MB) for the whole main kernel.
// ---------------------------------------------------------------------------
__global__ void prep_weights(const float* __restrict__ w1,
                             const float* __restrict__ w2,
                             _Float16* __restrict__ w1t,
                             _Float16* __restrict__ w2t) {
  int i = blockIdx.x * blockDim.x + threadIdx.x;
  if (i < HID_ * GP_) {
    int h = i / GP_, g = i % GP_;
    w1t[i] = (_Float16)((g < G_) ? w1[(size_t)g * HID_ + h] : 0.0f);
    return;
  }
  i -= HID_ * GP_;
  if (i < L_ * HID_) {
    int l = i / HID_, h = i % HID_;
    w2t[i] = (_Float16)w2[(size_t)h * L_ + l];
  }
}

__device__ __forceinline__ float sigmoidf_fast(float v) {
  return 1.0f / (1.0f + __expf(-v));
}

__device__ __forceinline__ v16h load_a16(const _Float16* p) {
  // 16-bit A-matrix 16x32 per-lane fragment: K[0..7] and K[16..23] (lo lanes),
  // shifted by +8 for hi lanes (caller bakes that into p).
  v8h a0 = *(const v8h*)(p);
  v8h a1 = *(const v8h*)(p + 16);
  return __builtin_shufflevector(a0, a1, 0, 1, 2, 3, 4, 5, 6, 7,
                                 8, 9, 10, 11, 12, 13, 14, 15);
}

// ---------------------------------------------------------------------------
// Fused kernel: one block = (batch b, 64 consecutive n). 256 threads = 8 waves.
// Each wave owns one 16-wide output-column tile and FOUR 16-row M-tiles, so
// every weight (B-operand) register load feeds four v_wmma issues. The GEMM1
// k-loop is fully unrolled so the scheduler can hoist the L2 weight stream
// several k-steps ahead of its consuming WMMA (rolled loops were observed to
// sink the loads into the consuming iteration).
// ---------------------------------------------------------------------------
__global__ __launch_bounds__(256) void fused_gate_mlp(
    const float* __restrict__ x,       // [B][L][N]
    const float* __restrict__ b1,      // [HID]
    const float* __restrict__ b2,      // [L]
    const int*   __restrict__ sel_idx, // [NSEL]
    const _Float16* __restrict__ w1t,  // [HID][GP]
    const _Float16* __restrict__ w2t,  // [L][HID]
    float* __restrict__ out)           // [B][L][N]
{
  __shared__ _Float16 sg[ROWS_ * GSTRIDE];  // gate tile, f16, row = n, col = g
  __shared__ _Float16 sh[ROWS_ * HSTRIDE];  // hidden tile, f16, row = n, col = h

  const int blk = blockIdx.x;
  const int b   = blk / (N_ / ROWS_);
  const int n0  = (blk % (N_ / ROWS_)) * ROWS_;
  const float* __restrict__ xb = x + (size_t)b * L_ * N_;

  const int t  = threadIdx.x;
  const int m  = t & (ROWS_ - 1);  // n-row within tile -> 256B coalesced reads
  const int gw = t / ROWS_;        // 4 parallel g-workers

  // ---- Build gate tile in LDS: [x1 (64) | xt (512) | sel (153) | pad (7)] ----
  for (int g = gw; g < GP_; g += 256 / ROWS_) {
    float v;
    if (g < CHUNK_) {
      v = xb[(size_t)(g * NCHNK_) * N_ + n0 + m];
    } else if (g < CHUNK_ + L_) {
      v = xb[(size_t)(g - CHUNK_) * N_ + n0 + m];
    } else if (g < G_) {
      int li = sel_idx[g - (CHUNK_ + L_)];
      int lj = (li + 1 < L_) ? li + 1 : L_ - 1;
      v = 0.75f * xb[(size_t)li * N_ + n0 + m] +
          0.25f * xb[(size_t)lj * N_ + n0 + m];
    } else {
      v = 0.0f;
    }
    sg[m * GSTRIDE + g] = (_Float16)v;
  }
  __syncthreads();

  const int wave = t >> 5;          // 0..7
  const int lane = t & 31;
  const int col  = lane & 15;       // A row / B,C,D column within 16x16 tile
  const int hiK  = (lane >> 4);     // lane-half selects K sub-range

  // ---- GEMM1: h[64 x 128] = relu(gate[64 x 729] @ w1 + b1) -------------------
  // wave -> HID columns [wave*16, wave*16+16), M-tiles 0..3.
  const int h0 = wave * 16;
  const float bias1 = b1[h0 + col];
  v8f acc[MT_];
#pragma unroll
  for (int mt = 0; mt < MT_; ++mt)
    acc[mt] = (v8f){bias1, bias1, bias1, bias1, bias1, bias1, bias1, bias1};
  {
    const _Float16* __restrict__ wrow = w1t + (size_t)(h0 + col) * GP_;
    const _Float16* __restrict__ arow = &sg[col * GSTRIDE + hiK * 8];
#pragma unroll
    for (int kt = 0; kt < KT1_; ++kt) {
      v16h bv = load_a16(wrow + kt * 32 + hiK * 8);  // feeds 4 WMMAs
      v16h a[MT_];
#pragma unroll
      for (int mt = 0; mt < MT_; ++mt)
        a[mt] = load_a16(arow + mt * 16 * GSTRIDE + kt * 32);
#pragma unroll
      for (int mt = 0; mt < MT_; ++mt)
        acc[mt] = __builtin_amdgcn_wmma_f32_16x16x32_f16(
            false, a[mt], false, bv, (short)0, acc[mt], false, false);
    }
  }
  // ReLU + stage hidden tile (D layout: VGPR r -> M = mt*16 + hiK*8 + r, N = col)
#pragma unroll
  for (int mt = 0; mt < MT_; ++mt) {
#pragma unroll
    for (int r = 0; r < 8; ++r) {
      float v = acc[mt][r] > 0.0f ? acc[mt][r] : 0.0f;
      sh[(mt * 16 + hiK * 8 + r) * HSTRIDE + h0 + col] = (_Float16)v;
    }
  }
  __syncthreads();

  // ---- GEMM2 + sigmoid + scale-by-x + store ---------------------------------
  // wave -> L columns [wave*64, wave*64+64) (4 tiles), M-tiles 0..3.
  const _Float16* __restrict__ arow2 = &sh[col * HSTRIDE + hiK * 8];
#pragma unroll
  for (int tile = 0; tile < 4; ++tile) {
    const int l0 = wave * 64 + tile * 16;
    const float bias2 = b2[l0 + col];
    v8f acc2[MT_];
#pragma unroll
    for (int mt = 0; mt < MT_; ++mt)
      acc2[mt] = (v8f){bias2, bias2, bias2, bias2, bias2, bias2, bias2, bias2};
    const _Float16* __restrict__ wrow2 = w2t + (size_t)(l0 + col) * HID_;
#pragma unroll
    for (int kt = 0; kt < HID_ / 32; ++kt) {
      v16h bv = load_a16(wrow2 + kt * 32 + hiK * 8);  // feeds 4 WMMAs
      v16h a[MT_];
#pragma unroll
      for (int mt = 0; mt < MT_; ++mt)
        a[mt] = load_a16(arow2 + mt * 16 * HSTRIDE + kt * 32);
#pragma unroll
      for (int mt = 0; mt < MT_; ++mt)
        acc2[mt] = __builtin_amdgcn_wmma_f32_16x16x32_f16(
            false, a[mt], false, bv, (short)0, acc2[mt], false, false);
    }
    // Epilogue: lane holds column l = l0+col; rows n0 + mt*16 + hiK*8 + (0..7).
    // Lane pair (l, l+16) covers 64 contiguous bytes per M-tile -> clean 64B txns,
    // 256B contiguous per out row across the 4 M-tiles.
    const int l = l0 + col;
    const float* __restrict__ xrowb = xb + (size_t)l * N_ + n0 + hiK * 8;
    float* __restrict__ orowb = out + ((size_t)b * L_ + l) * N_ + n0 + hiK * 8;
#pragma unroll
    for (int mt = 0; mt < MT_; ++mt) {
      const float* __restrict__ xrow = xrowb + mt * 16;
      float*       __restrict__ orow = orowb + mt * 16;
      float4 x0 = *(const float4*)(xrow);
      float4 x1 = *(const float4*)(xrow + 4);
      float4 o0, o1;
      o0.x = x0.x * sigmoidf_fast(acc2[mt][0]);
      o0.y = x0.y * sigmoidf_fast(acc2[mt][1]);
      o0.z = x0.z * sigmoidf_fast(acc2[mt][2]);
      o0.w = x0.w * sigmoidf_fast(acc2[mt][3]);
      o1.x = x1.x * sigmoidf_fast(acc2[mt][4]);
      o1.y = x1.y * sigmoidf_fast(acc2[mt][5]);
      o1.z = x1.z * sigmoidf_fast(acc2[mt][6]);
      o1.w = x1.w * sigmoidf_fast(acc2[mt][7]);
      *(float4*)(orow)     = o0;
      *(float4*)(orow + 4) = o1;
    }
  }
}

// ---------------------------------------------------------------------------
// Launcher. Inputs (setup_inputs order): x, w1, b1, w2, b2, sel_idx.
// d_ws layout: [0, 128*736*2) = w1t f16 ; next 512*128*2 = w2t f16 (~312 KB).
// ---------------------------------------------------------------------------
extern "C" void kernel_launch(void* const* d_in, const int* in_sizes, int n_in,
                              void* d_out, int out_size, void* d_ws, size_t ws_size,
                              hipStream_t stream) {
  const float* x   = (const float*)d_in[0];
  const float* w1  = (const float*)d_in[1];
  const float* b1  = (const float*)d_in[2];
  const float* w2  = (const float*)d_in[3];
  const float* b2  = (const float*)d_in[4];
  const int*   sel = (const int*)d_in[5];

  _Float16* w1t = (_Float16*)d_ws;
  _Float16* w2t = (_Float16*)((char*)d_ws + (size_t)HID_ * GP_ * sizeof(_Float16));
  float*    out = (float*)d_out;

  int prep_elems = HID_ * GP_ + L_ * HID_;
  prep_weights<<<(prep_elems + 255) / 256, 256, 0, stream>>>(w1, w2, w1t, w2t);

  dim3 grid(B_ * (N_ / ROWS_));   // 1024 blocks
  fused_gate_mlp<<<grid, 256, 0, stream>>>(x, b1, b2, sel, w1t, w2t, out);
}
